// TimeLSTM_v1_2491081032090
// MI455X (gfx1250) — matrix operand order repaired and checked by reading, not compiled, not run on hardware
//
#include <hip/hip_runtime.h>
#include <hip/hip_bf16.h>

// ---------------------------------------------------------------------------
// Fused TimeLSTM for MI455X (gfx1250, wave32, WMMA).
//
//   B = 512 batch rows, T = 512 steps, F = 128 features.
//   Each workgroup owns a 16-row batch slab for the whole time loop
//   (recurrence is independent per batch row). Per step, per gate:
//       gate = act( [h | x_t] (16x256, f16 in LDS)  @  [W_h ; W_x] (256x128) + b )
//   done with v_wmma_f32_16x16x32_f16, K=256 => 8 k-chunks.
//   16 waves per WG: wave = (gate g in 0..3, n-quarter nq in 0..3),
//   each wave computes a 16x32 slice = 2 N-tiles x 8 K-chunks = 16 WMMAs/step,
//   with all its B fragments resident in registers for all 512 steps.
//   x_{t+1} global load is software-pipelined behind the GEMM.
//   Activations are branchless: act(v) = a*sigmoid(s*v)+b with hardware rcp.
// ---------------------------------------------------------------------------

typedef __attribute__((ext_vector_type(16))) _Float16 v16h;
typedef __attribute__((ext_vector_type(8)))  _Float16 v8h;
typedef __attribute__((ext_vector_type(4)))  _Float16 v4h;
typedef __attribute__((ext_vector_type(8)))  float    v8f;

#define LSTM_B 512
#define LSTM_T 512
#define LSTM_F 128

// ---------------------------------------------------------------------------
// Prep kernel: convert the 8 f32 weight matrices (4x W_h [128x128], 4x W_x
// [128x128], both stored [k][n] row-major) into f16 B-matrix fragments in the
// contiguous k order this file uses for both WMMA operands (16x16x32):
//   lane L holds column n = L%16; halfs h=0..15 hold k = (L/16)*16 + h.
// Fragment index: (g, nt, kc) with kc 0..3 -> W_h k-chunks, kc 4..7 -> W_x.
// Each fragment = 32 lanes * 16 halfs = 512 halfs. Total 4*8*8*512 halfs = 256KB.
// ---------------------------------------------------------------------------
__global__ void lstm_prep_frags(const float* __restrict__ wh0, const float* __restrict__ wh1,
                                const float* __restrict__ wh2, const float* __restrict__ wh3,
                                const float* __restrict__ wx0, const float* __restrict__ wx1,
                                const float* __restrict__ wx2, const float* __restrict__ wx3,
                                _Float16* __restrict__ frags) {
    int idx  = blockIdx.x * blockDim.x + threadIdx.x;   // one half each; 131072 total
    int h    = idx & 15;
    int lane = (idx >> 4) & 31;
    int kc   = (idx >> 9) & 7;
    int nt   = (idx >> 12) & 7;
    int g    = (idx >> 15) & 3;
    const float* W;
    if (kc < 4) W = (g == 0) ? wh0 : (g == 1) ? wh1 : (g == 2) ? wh2 : wh3;
    else        W = (g == 0) ? wx0 : (g == 1) ? wx1 : (g == 2) ? wx2 : wx3;
    int k = ((kc & 3) * 32) + ((lane >> 4) * 16) + h;   // 0..127 within the matrix
    int n = nt * 16 + (lane & 15);
    frags[idx] = (_Float16)W[k * LSTM_F + n];
}

// Raw-hardware-rcp sigmoid/tanh: v_exp_f32 + v_rcp_f32, no IEEE div expansion.
__device__ __forceinline__ float fast_sigmoid(float x) {
    return __builtin_amdgcn_rcpf(1.0f + __expf(-x));
}
__device__ __forceinline__ float fast_tanh(float x) {
    // tanh(x) = 2*sigmoid(2x) - 1; saturates correctly for large |x|.
    return 2.0f * __builtin_amdgcn_rcpf(1.0f + __expf(-2.0f * x)) - 1.0f;
}

// ---------------------------------------------------------------------------
// Main recurrent kernel. Grid = 32 workgroups x 512 threads (16 waves).
// ---------------------------------------------------------------------------
__launch_bounds__(512, 1)
__global__ void lstm_scan(const float* __restrict__ x,
                          const float* __restrict__ b0, const float* __restrict__ b1,
                          const float* __restrict__ b2, const float* __restrict__ b3,
                          const _Float16* __restrict__ frags,
                          float* __restrict__ out) {
    // A-tile: 16 rows x 256 cols f16. cols [0,128) = h, cols [128,256) = x_t.
    __shared__ _Float16 a_lds[16][256];          // 8 KB
    __shared__ float    gates_lds[4][16][128];   // 32 KB

    const int tid  = threadIdx.x;
    const int lane = tid & 31;
    const int wave = tid >> 5;        // 0..15
    const int g    = wave >> 2;       // gate: 0=ig 1=fg 2=in(tanh) 3=og
    const int nq   = wave & 3;        // n-quarter: cols [nq*32, nq*32+32)
    const int slab = blockIdx.x * 16; // batch rows [slab, slab+16)

    // --- Load this wave's 16 B-fragments into registers (resident all steps).
    v16h bfr[2][8];
#pragma unroll
    for (int nt2 = 0; nt2 < 2; ++nt2) {
#pragma unroll
        for (int kc = 0; kc < 8; ++kc) {
            const int fidx = ((g * 8 + (nq * 2 + nt2)) * 8 + kc);
            bfr[nt2][kc] = *(const v16h*)(frags + (size_t)fidx * 512 + lane * 16);
        }
    }

    // --- Per-lane bias (C-matrix column n = L%16 within each N-tile).
    const float* biasg = (g == 0) ? b0 : (g == 1) ? b1 : (g == 2) ? b2 : b3;
    const float bias_v0 = biasg[nq * 32 + (lane & 15)];
    const float bias_v1 = biasg[nq * 32 + 16 + (lane & 15)];

    // --- Branchless activation constants for this wave's gate:
    //     act(v) = act_a * sigmoid(act_s * v) + act_b
    //     sigmoid gates: (1,1,0); tanh gate (g==2): (2,2,-1).
    const float act_s = (g == 2) ? 2.0f : 1.0f;
    const float act_a = act_s;
    const float act_b = (g == 2) ? -1.0f : 0.0f;

    // --- Combine-stage element mapping: thread owns 4 elements e = tid + 512*j
    //     of the flattened 16x128 slab; c state stays in registers.
    float c[4] = {0.f, 0.f, 0.f, 0.f};

    // --- x load mapping: one global_load_b128 per thread per step.
    const int xrow  = tid >> 5;         // 0..15
    const int xcol4 = (tid & 31) * 4;   // 0..124

    // --- Init h = 0 in LDS.
#pragma unroll
    for (int j = 0; j < 4; ++j) {
        const int e = tid + 512 * j;
        a_lds[e >> 7][e & 127] = (_Float16)0.0f;
    }
    __syncthreads();

    const size_t xrow_base = ((size_t)(slab + xrow) * LSTM_T) * LSTM_F + xcol4;

    // Software pipeline: preload x_0.
    float4 xv = *(const float4*)(x + xrow_base);

    for (int t = 0; t < LSTM_T; ++t) {
        // ---- Stage x_t (preloaded) into the A-tile (f32 -> f16). -----------
        v4h xh;
        xh[0] = (_Float16)xv.x; xh[1] = (_Float16)xv.y;
        xh[2] = (_Float16)xv.z; xh[3] = (_Float16)xv.w;
        *(v4h*)&a_lds[xrow][128 + xcol4] = xh;   // ds_write_b64
        __syncthreads();                          // [h | x_t] tile ready

        // ---- Issue x_{t+1} load now; it overlaps the 16-WMMA GEMM below. ---
        if (t + 1 < LSTM_T)
            xv = *(const float4*)(x + xrow_base + (size_t)(t + 1) * LSTM_F);

        // ---- GEMM: this wave's 16x32 gate slice, K = 256. ------------------
        v8f acc0 = {bias_v0, bias_v0, bias_v0, bias_v0, bias_v0, bias_v0, bias_v0, bias_v0};
        v8f acc1 = {bias_v1, bias_v1, bias_v1, bias_v1, bias_v1, bias_v1, bias_v1, bias_v1};
#pragma unroll
        for (int kc = 0; kc < 8; ++kc) {
            // A fragment (contiguous k order, same as B frags), 16x32 tile: lane holds row m = L%16;
            // halfs 0..15 -> k = kc*32 + (L/16)*16 + h.
            const int m  = lane & 15;
            const int kb = kc * 32 + (lane >> 4) * 16;
            const v8h alo = *(const v8h*)&a_lds[m][kb];        // ds_load_b128
            const v8h ahi = *(const v8h*)&a_lds[m][kb + 8];    // ds_load_b128
            v16h a;
#pragma unroll
            for (int i = 0; i < 8; ++i) { a[i] = alo[i]; a[8 + i] = ahi[i]; }
            acc0 = __builtin_amdgcn_wmma_f32_16x16x32_f16(false, a, false, bfr[0][kc],
                                                          (short)0, acc0, false, false);
            acc1 = __builtin_amdgcn_wmma_f32_16x16x32_f16(false, a, false, bfr[1][kc],
                                                          (short)0, acc1, false, false);
        }

        // ---- Branchless activation + scatter to gates_lds. -----------------
        // D layout: VGPR r -> row m = r + 8*(L/16); col n = tile_n + L%16.
#pragma unroll
        for (int r = 0; r < 8; ++r) {
            const float v0 = act_a * fast_sigmoid(act_s * acc0[r]) + act_b;
            const float v1 = act_a * fast_sigmoid(act_s * acc1[r]) + act_b;
            const int m = r + (lane >> 4) * 8;
            gates_lds[g][m][nq * 32 +      (lane & 15)] = v0;
            gates_lds[g][m][nq * 32 + 16 + (lane & 15)] = v1;
        }
        __syncthreads();                          // all 4 gates ready

        // ---- Elementwise combine: c = fg*c + ig*gt ; h = og*tanh(c). -------
#pragma unroll
        for (int j = 0; j < 4; ++j) {
            const int e = tid + 512 * j;
            const int m = e >> 7, n = e & 127;
            const float ig = gates_lds[0][m][n];
            const float fg = gates_lds[1][m][n];
            const float gt = gates_lds[2][m][n];
            const float og = gates_lds[3][m][n];
            c[j] = fg * c[j] + ig * gt;
            const float hval = og * fast_tanh(c[j]);
            a_lds[m][n] = (_Float16)hval;         // h for next step's A-tile
            if (t == LSTM_T - 1)
                out[(size_t)(slab + m) * LSTM_F + n] = hval;
        }
        // next iteration's __syncthreads (after x write) orders h writes
        // against the next A-tile reads.
    }
}

extern "C" void kernel_launch(void* const* d_in, const int* in_sizes, int n_in,
                              void* d_out, int out_size, void* d_ws, size_t ws_size,
                              hipStream_t stream) {
    const float* x = (const float*)d_in[0];
    const float* wh[4]; const float* wx[4]; const float* bias[4];
    for (int g = 0; g < 4; ++g) {
        wh[g]   = (const float*)d_in[1 + 3 * g];  // *_w_h [128,128] (k-major)
        wx[g]   = (const float*)d_in[2 + 3 * g];  // *_w_x [128,128] (k-major)
        bias[g] = (const float*)d_in[3 + 3 * g];  // *_b   [128]
    }
    _Float16* frags = (_Float16*)d_ws;            // 256 KB of scratch used

    // 131072 fragment halfs -> 512 blocks x 256 threads
    lstm_prep_frags<<<512, 256, 0, stream>>>(wh[0], wh[1], wh[2], wh[3],
                                             wx[0], wx[1], wx[2], wx[3], frags);

    // 512 batch rows / 16-row slabs = 32 workgroups x 512 threads
    lstm_scan<<<32, 512, 0, stream>>>(x, bias[0], bias[1], bias[2], bias[3],
                                      frags, (float*)d_out);
}
